// GraphTransposeDecoderBlock_10273561772256
// MI455X (gfx1250) — matrix operand order repaired and checked by reading, not compiled, not run on hardware
//
#include <hip/hip_runtime.h>
#include <hip/hip_bf16.h>

typedef __attribute__((ext_vector_type(16))) _Float16 v16h;
typedef __attribute__((ext_vector_type(8)))  _Float16 v8h;
typedef __attribute__((ext_vector_type(8)))  float    v8f;

#define DD     128
#define LNEPS  1e-5f
#define LDSPAD 8          // halfwords of row padding -> 272B stride, bank-rotating

__device__ __forceinline__ v16h cat8(v8h lo, v8h hi) {
    return __builtin_shufflevector(lo, hi, 0,1,2,3,4,5,6,7,8,9,10,11,12,13,14,15);
}

// ------------------------------------------------------------------
// Kernel 1: LayerNorm + (1+eps) pre-scale.  One wave32 per row.
//   h[row]   = LN(x[row]) * gamma + beta
//   acc[row] = (1+eps) * h[row]
// ------------------------------------------------------------------
__global__ __launch_bounds__(256)
void ln_prescale_kernel(const float* __restrict__ x,
                        const float* __restrict__ gamma,
                        const float* __restrict__ beta,
                        const float* __restrict__ eps_p,
                        float* __restrict__ h,
                        float* __restrict__ acc,
                        int n)
{
    const int wave = threadIdx.x >> 5;
    const int lane = threadIdx.x & 31;
    const int row  = blockIdx.x * 8 + wave;
    if (row >= n) return;

    const float4 v = ((const float4*)(x + (size_t)row * DD))[lane];
    float s = v.x + v.y + v.z + v.w;
    float q = v.x*v.x + v.y*v.y + v.z*v.z + v.w*v.w;
    #pragma unroll
    for (int m = 16; m > 0; m >>= 1) {          // wave32 reduction
        s += __shfl_xor(s, m, 32);
        q += __shfl_xor(q, m, 32);
    }
    const float mu  = s * (1.0f / DD);
    const float var = q * (1.0f / DD) - mu * mu;
    const float inv = rsqrtf(var + LNEPS);
    const float sc  = 1.0f + eps_p[0];

    const float4 g = ((const float4*)gamma)[lane];
    const float4 b = ((const float4*)beta )[lane];
    float4 hv;
    hv.x = (v.x - mu) * inv * g.x + b.x;
    hv.y = (v.y - mu) * inv * g.y + b.y;
    hv.z = (v.z - mu) * inv * g.z + b.z;
    hv.w = (v.w - mu) * inv * g.w + b.w;

    ((float4*)(h   + (size_t)row * DD))[lane] = hv;
    float4 av; av.x = sc*hv.x; av.y = sc*hv.y; av.z = sc*hv.z; av.w = sc*hv.w;
    ((float4*)(acc + (size_t)row * DD))[lane] = av;
}

// ------------------------------------------------------------------
// Kernel 2: segment_sum scatter.  One wave per edge: 32 lanes x float4.
// acc[rows[e]] += h[cols[e]]  (f32 atomics resolve in the 192MB L2;
// the whole 25.6MB accumulator is L2-resident)
// ------------------------------------------------------------------
__global__ __launch_bounds__(256)
void edge_scatter_kernel(const float* __restrict__ h,
                         const int* __restrict__ rows,
                         const int* __restrict__ cols,
                         float* __restrict__ acc,
                         int e)
{
    const long tid = (long)blockIdx.x * blockDim.x + threadIdx.x;
    if (tid >= (long)e * 32) return;
    const int edge = (int)(tid >> 5);
    const int part = (int)(tid & 31);
    const int src  = cols[edge];           // wave-uniform -> scalar broadcast
    const int dst  = rows[edge];
    const float4 v = *(const float4*)(h + (size_t)src * DD + part * 4);
    float* d = acc + (size_t)dst * DD + part * 4;
    atomicAdd(d + 0, v.x);
    atomicAdd(d + 1, v.y);
    atomicAdd(d + 2, v.z);
    atomicAdd(d + 3, v.w);
}

// ------------------------------------------------------------------
// Kernel 3/4: out = epilogue(A @ W^T + bias), W is [128,128] row-major.
// 128 threads = 4 wave32; each wave owns a 16-row M-tile; W (f16) and a
// 64x128 A-tile live in padded LDS.  v_wmma_f32_16x16x32_f16, f32 accum.
// A fragments (invariant over the N-tile loop) are hoisted into 32 VGPRs;
// the inner loop is then B-fragment ds_load_b128 pairs + WMMA only.
// Templated on SILU / RESID so the unrolled epilogue has no branches.
//
// Fragment packing per CDNA5 ISA 7.12.2 (wave32):
//   A 16x32 f16:  lane L -> row M=L%16; half=L/16 selects K-pairs
//                 VGPR0..3 = K  half*8 + 0..7, VGPR4..7 = K 16+half*8+0..7
//                 -> two contiguous 8-halfword runs  (two ds_load_b128)
//   B 32x16 f16:  lane L -> col N=L%16; same K order as A:
//                 K half*8+0..7, then 16+half*8+0..7; B(k,n) = W[n][k]
//   C/D 16x16 f32: VGPR v, half -> M = v + 8*half, N = L%16
// ------------------------------------------------------------------
template <bool SILU, bool RESID>
__global__ __launch_bounds__(128)
void gemm_epilogue_kernel(const float* __restrict__ A,
                          const float* __restrict__ W,
                          const float* __restrict__ bias,
                          const float* __restrict__ resid,
                          float* __restrict__ out,
                          int n)
{
    __shared__ _Float16 sW[DD][DD + LDSPAD];   // sW[n][k] = W[n][k]
    __shared__ _Float16 sA[64][DD + LDSPAD];

    const int t  = threadIdx.x;
    const int m0 = blockIdx.x * 64;

    // Stage W as f16 (coalesced float4 reads)
    for (int idx = t * 4; idx < DD * DD; idx += 128 * 4) {
        const float4 w4 = *(const float4*)(W + idx);
        const int r = idx >> 7, k = idx & (DD - 1);
        sW[r][k+0] = (_Float16)w4.x;  sW[r][k+1] = (_Float16)w4.y;
        sW[r][k+2] = (_Float16)w4.z;  sW[r][k+3] = (_Float16)w4.w;
    }
    // Stage 64-row A tile as f16 (zero-fill past n)
    for (int idx = t * 4; idx < 64 * DD; idx += 128 * 4) {
        const int r = idx >> 7, k = idx & (DD - 1);
        const int g = m0 + r;
        float4 a4 = make_float4(0.f, 0.f, 0.f, 0.f);
        if (g < n) a4 = *(const float4*)(A + (size_t)g * DD + k);
        sA[r][k+0] = (_Float16)a4.x;  sA[r][k+1] = (_Float16)a4.y;
        sA[r][k+2] = (_Float16)a4.z;  sA[r][k+3] = (_Float16)a4.w;
    }
    __syncthreads();

    const int wv    = t >> 5;
    const int lane  = t & 31;
    const int half  = lane >> 4;
    const int lr    = lane & 15;
    const int mbase = m0 + wv * 16;
    if (mbase >= n) return;                   // wave-uniform: EXEC stays all-1s

    // Hoist the 4 A fragments (invariant over nt): 32 VGPRs
    v16h afrag[4];
    #pragma unroll
    for (int kt = 0; kt < 4; ++kt) {
        const int ra = wv * 16 + lr;
        const int ka = kt * 32 + half * 8;
        afrag[kt] = cat8(*(const v8h*)&sA[ra][ka],
                         *(const v8h*)&sA[ra][ka + 16]);
    }

    #pragma unroll
    for (int nt = 0; nt < 8; ++nt) {          // 8 N-tiles of 16 cols
        v8f c = {0.f, 0.f, 0.f, 0.f, 0.f, 0.f, 0.f, 0.f};
        #pragma unroll
        for (int kt = 0; kt < 4; ++kt) {      // K = 128 = 4 x 32
            const int nb = nt * 16 + lr;
            const int kb = kt * 32 + half * 8;
            const v16h b = cat8(*(const v8h*)&sW[nb][kb],
                                *(const v8h*)&sW[nb][kb + 16]);

            c = __builtin_amdgcn_wmma_f32_16x16x32_f16(
                    /*neg_a=*/false, afrag[kt], /*neg_b=*/false, b,
                    /*c_mod=*/(short)0, c, /*reuse_a=*/false, /*reuse_b=*/false);
        }
        const int   col = nt * 16 + lr;
        const float bv  = bias[col];
        #pragma unroll
        for (int v = 0; v < 8; ++v) {
            const int row = mbase + half * 8 + v;
            float val = c[v] + bv;
            if (SILU) {
                // silu(x) = x * sigmoid(x); fast path: v_exp_f32 + v_rcp_f32
                const float sig = __builtin_amdgcn_rcpf(1.0f + __expf(-val));
                val *= sig;
            }
            if (RESID) val += resid[(size_t)row * DD + col];
            out[(size_t)row * DD + col] = val;
        }
    }
}

// ------------------------------------------------------------------
extern "C" void kernel_launch(void* const* d_in, const int* in_sizes, int n_in,
                              void* d_out, int out_size, void* d_ws, size_t ws_size,
                              hipStream_t stream)
{
    const float* x     = (const float*)d_in[0];
    const int*   erows = (const int*)  d_in[1];
    const int*   ecols = (const int*)  d_in[2];
    const float* W1    = (const float*)d_in[3];
    const float* b1    = (const float*)d_in[4];
    const float* W2    = (const float*)d_in[5];
    const float* b2    = (const float*)d_in[6];
    const float* gamma = (const float*)d_in[7];
    const float* beta  = (const float*)d_in[8];
    const float* eps   = (const float*)d_in[9];

    const int n = in_sizes[0] / DD;     // 50000
    const int e = in_sizes[1];          // 800000

    float* h    = (float*)d_ws;                 // [n,128] LN output
    float* acc  = h + (size_t)n * DD;           // [n,128] (1+eps)h + scatter
    float* tbuf = h;                            // lin1 output reuses h's buffer
    float* out  = (float*)d_out;

    // 1) LayerNorm + prescale (writes h and acc)
    ln_prescale_kernel<<<(n + 7) / 8, 256, 0, stream>>>(x, gamma, beta, eps, h, acc, n);

    // 2) segment_sum scatter-add into acc
    const long tot = (long)e * 32;
    edge_scatter_kernel<<<(unsigned)((tot + 255) / 256), 256, 0, stream>>>(h, erows, ecols, acc, e);

    // 3) t = silu(acc @ W1^T + b1)
    gemm_epilogue_kernel<true, false><<<(n + 63) / 64, 128, 0, stream>>>(acc, W1, b1, nullptr, tbuf, n);

    // 4) out = x + (t @ W2^T + b2)
    gemm_epilogue_kernel<false, true><<<(n + 63) / 64, 128, 0, stream>>>(tbuf, W2, b2, x, out, n);
}
